// CustomGatedGraphConv_77129022701746
// MI455X (gfx1250) — matrix operand order repaired and checked by reading, not compiled, not run on hardware
//
#include <hip/hip_runtime.h>
#include <hip/hip_bf16.h>
#include <stdint.h>

// ---------------------------------------------------------------------------
// GatedGraphConv on MI455X (gfx1250, wave32):
//   per layer: m = h @ W[l]                         (bf16 WMMA, fp32 accum)
//              agg = scatter_add(m[src]*attr, dst)  (global_atomic_add_f32)
//              h = GRU(agg, h)                      (fused 6-accumulator WMMA)
// Activations kept as fp32 master + bf16 shadow (WMMA operands), so inner
// GEMM loops are pure b128-load + wmma with no cvt chains.
// ---------------------------------------------------------------------------

#define N_NODES   20000
#define N_EDGES   320000
#define D         256
#define D3        768
#define NUM_LAYERS 5

typedef __attribute__((ext_vector_type(16))) __bf16 v16bf;
typedef __attribute__((ext_vector_type(8)))  float  v8f;

// ---- fragment loaders (wave32; contiguous K order k = 16*half + s in A and B) ------------------

// A-matrix (16x32 bf16): lane r=lane&15 is row M=r, half=lane>>4 selects
// K run [kc+half*16 .. +15] (matches B fragment). Two 16B loads.
__device__ __forceinline__ v16bf load_a_frag(const __bf16* __restrict__ rowp,
                                             int kc, int half) {
  union { uint4 u[2]; v16bf v; } cv;
  cv.u[0] = *reinterpret_cast<const uint4*>(rowp + kc + half * 16);
  cv.u[1] = *reinterpret_cast<const uint4*>(rowp + kc + half * 16 + 8);
  return cv.v;
}

// B-matrix (32x16 bf16): lane r = column N, half selects K offset (0 / 16);
// weights stored column-major so 16 K-values are contiguous (32 B).
__device__ __forceinline__ v16bf load_b_frag(const __bf16* __restrict__ colp, int koff) {
  union { uint4 u[2]; v16bf v; } cv;
  const uint4* p = reinterpret_cast<const uint4*>(colp + koff);
  cv.u[0] = p[0];
  cv.u[1] = p[1];
  return cv.v;
}

__device__ __forceinline__ v8f wmma_bf16(v16bf a, v16bf b, v8f c) {
  return __builtin_amdgcn_wmma_f32_16x16x32_bf16(false, a, false, b, (short)0, c,
                                                 false, false);
}

__device__ __forceinline__ float sigmoidf(float x) {
  return 1.0f / (1.0f + __expf(-x));
}

// ---- weight packing --------------------------------------------------------
// Wt[l][n][k] = W[l][k][n]   (5 x 256 x 256 bf16, column-major B layout)
// wihb / whhb  = bf16 copies of w_ih / w_hh ([768][256], already B-layout)
__global__ void pack_weights_kernel(const float* __restrict__ W,
                                    const float* __restrict__ wih,
                                    const float* __restrict__ whh,
                                    __bf16* __restrict__ Wt,
                                    __bf16* __restrict__ wihb,
                                    __bf16* __restrict__ whhb) {
  int idx = blockIdx.x * blockDim.x + threadIdx.x;
  const int nWt = NUM_LAYERS * D * D;   // 327680
  const int nG  = D3 * D;               // 196608
  if (idx < nWt) {
    int l = idx >> 16;
    int c = (idx >> 8) & 255;
    int k = idx & 255;
    Wt[idx] = (__bf16)W[l * (D * D) + k * D + c];
    return;
  }
  int i2 = idx - nWt;
  if (i2 < nG) { wihb[i2] = (__bf16)wih[i2]; return; }
  int i3 = i2 - nG;
  if (i3 < nG) { whhb[i3] = (__bf16)whh[i3]; }
}

// ---- fp32 -> bf16 shadow conversion (8 elems / thread, 16B stores) ---------
__global__ void cvt_bf16_kernel(const float* __restrict__ src,
                                __bf16* __restrict__ dst, int n8) {
  int i = blockIdx.x * blockDim.x + threadIdx.x;
  if (i >= n8) return;
  const float4* p = reinterpret_cast<const float4*>(src) + (size_t)i * 2;
  float4 a = p[0], b = p[1];
  union { uint4 u; __bf16 v[8]; } cv;
  cv.v[0] = (__bf16)a.x; cv.v[1] = (__bf16)a.y; cv.v[2] = (__bf16)a.z; cv.v[3] = (__bf16)a.w;
  cv.v[4] = (__bf16)b.x; cv.v[5] = (__bf16)b.y; cv.v[6] = (__bf16)b.z; cv.v[7] = (__bf16)b.w;
  reinterpret_cast<uint4*>(dst)[i] = cv.u;
}

// ---- m = h @ W[l]: one wave -> 16x64 tile (4 accumulators share A-frag) ----
__global__ __launch_bounds__(256) void gemm_msg_kernel(
    const __bf16* __restrict__ hbf, const __bf16* __restrict__ Wt,
    float* __restrict__ m) {
  int wave = (blockIdx.x * blockDim.x + threadIdx.x) >> 5;  // 0..4999
  int lane = threadIdx.x & 31;
  int tile_m = wave >> 2;   // 0..1249
  int nq     = wave & 3;    // 64-wide column quadrant
  int r = lane & 15, half = lane >> 4;
  const __bf16* arow = hbf + (size_t)(tile_m * 16 + r) * D;
  const __bf16* b0 = Wt + (size_t)(nq * 64 +  0 + r) * D;
  const __bf16* b1 = Wt + (size_t)(nq * 64 + 16 + r) * D;
  const __bf16* b2 = Wt + (size_t)(nq * 64 + 32 + r) * D;
  const __bf16* b3 = Wt + (size_t)(nq * 64 + 48 + r) * D;
  v8f acc0 = {}, acc1 = {}, acc2 = {}, acc3 = {};
#pragma unroll
  for (int kc = 0; kc < D; kc += 32) {
    v16bf a = load_a_frag(arow, kc, half);
    int off = kc + half * 16;
    acc0 = wmma_bf16(a, load_b_frag(b0, off), acc0);
    acc1 = wmma_bf16(a, load_b_frag(b1, off), acc1);
    acc2 = wmma_bf16(a, load_b_frag(b2, off), acc2);
    acc3 = wmma_bf16(a, load_b_frag(b3, off), acc3);
  }
  // D layout: VGPR g -> row M = half*8 + g, col N = r
  float* mb = m + (size_t)(tile_m * 16 + half * 8) * D + nq * 64 + r;
#pragma unroll
  for (int g = 0; g < 8; ++g) {
    mb[(size_t)g * D +  0] = acc0[g];
    mb[(size_t)g * D + 16] = acc1[g];
    mb[(size_t)g * D + 32] = acc2[g];
    mb[(size_t)g * D + 48] = acc3[g];
  }
}

// ---- zero agg --------------------------------------------------------------
__global__ void zero_kernel(float4* __restrict__ p, int n4) {
  int i = blockIdx.x * blockDim.x + threadIdx.x;
  if (i < n4) p[i] = make_float4(0.f, 0.f, 0.f, 0.f);
}

// ---- edge gather * attr -> atomic scatter-add ------------------------------
__global__ __launch_bounds__(256) void edge_scatter_kernel(
    const float* __restrict__ m, const long long* __restrict__ src,
    const long long* __restrict__ dst, const float* __restrict__ attr,
    float* __restrict__ agg) {
  int wave = (blockIdx.x * blockDim.x + threadIdx.x) >> 5;
  int lane = threadIdx.x & 31;
  if (wave >= N_EDGES) return;
  int s = (int)src[wave];
  int d = (int)dst[wave];
  float w = attr[wave];
  const float4* mp = reinterpret_cast<const float4*>(m + (size_t)s * D) + lane * 2;
  float4 v0 = mp[0], v1 = mp[1];
  float* ap = agg + (size_t)d * D + lane * 8;
  atomicAdd(ap + 0, v0.x * w);
  atomicAdd(ap + 1, v0.y * w);
  atomicAdd(ap + 2, v0.z * w);
  atomicAdd(ap + 3, v0.w * w);
  atomicAdd(ap + 4, v1.x * w);
  atomicAdd(ap + 5, v1.y * w);
  atomicAdd(ap + 6, v1.z * w);
  atomicAdd(ap + 7, v1.w * w);
}

// ---- fused GRU: 6 WMMA accumulators (r/z/n x {agg path, h path}) -----------
// Reads bf16 shadows for WMMA, fp32 h for the gate blend; writes fp32 h' and
// its bf16 shadow for the next layer.
__global__ __launch_bounds__(256) void gru_kernel(
    const __bf16* __restrict__ aggbf, const __bf16* __restrict__ hbf,
    const float* __restrict__ h,
    const __bf16* __restrict__ wihb, const __bf16* __restrict__ whhb,
    const float* __restrict__ b_ih, const float* __restrict__ b_hh,
    float* __restrict__ hout, __bf16* __restrict__ houtbf) {
  int wave = (blockIdx.x * blockDim.x + threadIdx.x) >> 5;
  int lane = threadIdx.x & 31;
  int tile_m = wave >> 4;
  int tile_n = wave & 15;
  int r = lane & 15, half = lane >> 4;
  int col = tile_n * 16 + r;

  const __bf16* arow = aggbf + (size_t)(tile_m * 16 + r) * D;
  const __bf16* hrow = hbf   + (size_t)(tile_m * 16 + r) * D;
  const __bf16* cir = wihb + (size_t)col * D;            // gate r, input path
  const __bf16* ciz = wihb + (size_t)(D  + col) * D;     // gate z
  const __bf16* cin = wihb + (size_t)(2*D + col) * D;    // gate n
  const __bf16* chr = whhb + (size_t)col * D;            // hidden path
  const __bf16* chz = whhb + (size_t)(D  + col) * D;
  const __bf16* chn = whhb + (size_t)(2*D + col) * D;

  v8f air = {}, aiz = {}, ain = {}, ahr = {}, ahz = {}, ahn = {};
#pragma unroll 2
  for (int kc = 0; kc < D; kc += 32) {
    v16bf am = load_a_frag(arow, kc, half);
    v16bf ah = load_a_frag(hrow, kc, half);
    int off = kc + half * 16;
    air = wmma_bf16(am, load_b_frag(cir, off), air);
    aiz = wmma_bf16(am, load_b_frag(ciz, off), aiz);
    ain = wmma_bf16(am, load_b_frag(cin, off), ain);
    ahr = wmma_bf16(ah, load_b_frag(chr, off), ahr);
    ahz = wmma_bf16(ah, load_b_frag(chz, off), ahz);
    ahn = wmma_bf16(ah, load_b_frag(chn, off), ahn);
  }

  float bihr = b_ih[col], bihz = b_ih[D + col], bihn = b_ih[2 * D + col];
  float bhhr = b_hh[col], bhhz = b_hh[D + col], bhhn = b_hh[2 * D + col];

  int orow0 = tile_m * 16 + half * 8;
#pragma unroll
  for (int g = 0; g < 8; ++g) {
    int orow = orow0 + g;
    float rg = sigmoidf(air[g] + bihr + ahr[g] + bhhr);
    float zg = sigmoidf(aiz[g] + bihz + ahz[g] + bhhz);
    float ng = tanhf(ain[g] + bihn + rg * (ahn[g] + bhhn));
    float hv = h[(size_t)orow * D + col];
    float hn2 = (1.0f - zg) * ng + zg * hv;
    hout[(size_t)orow * D + col]   = hn2;
    houtbf[(size_t)orow * D + col] = (__bf16)hn2;
  }
}

// ---------------------------------------------------------------------------
extern "C" void kernel_launch(void* const* d_in, const int* in_sizes, int n_in,
                              void* d_out, int out_size, void* d_ws, size_t ws_size,
                              hipStream_t stream) {
  const float*      x     = (const float*)d_in[0];
  const long long*  eidx  = (const long long*)d_in[1];  // int64 per reference
  const float*      eattr = (const float*)d_in[2];
  const float*      W     = (const float*)d_in[3];
  const float*      wih   = (const float*)d_in[4];
  const float*      whh   = (const float*)d_in[5];
  const float*      bih   = (const float*)d_in[6];
  const float*      bhh   = (const float*)d_in[7];
  float*            out   = (float*)d_out;

  // workspace layout (all offsets 256B-aligned); total ~89.3 MB
  char* ws = (char*)d_ws;
  __bf16* Wt    = (__bf16*)(ws + 0);          // 655360
  __bf16* wihb  = (__bf16*)(ws + 655360);     // 393216
  __bf16* whhb  = (__bf16*)(ws + 1048576);    // 393216
  float*  m     = (float*)(ws + 1441792);     // 20480000
  float*  agg   = (float*)(ws + 21921792);    // 20480000
  float*  htmp  = (float*)(ws + 42401792);    // 20480000
  __bf16* aggbf = (__bf16*)(ws + 62881792);   // 10240000
  __bf16* hbfA  = (__bf16*)(ws + 73121792);   // 10240000
  __bf16* hbfB  = (__bf16*)(ws + 83361792);   // 10240000

  const long long* src = eidx;
  const long long* dst = eidx + N_EDGES;

  pack_weights_kernel<<<2816, 256, 0, stream>>>(W, wih, whh, Wt, wihb, whhb);

  const int n8          = N_NODES * D / 8;              // 640,000
  const int cvt_blocks  = (n8 + 255) / 256;             // 2500
  const int gemm_blocks = (N_NODES / 16) * 4 / 8;       // 625 (5000 waves)
  const int gru_blocks  = (N_NODES / 16) * (D / 16) / 8;// 2500 (20000 waves)
  const int edge_blocks = N_EDGES / 8;                  // 40000
  const int n4          = N_NODES * D / 4;              // 1,280,000
  const int zero_blocks = (n4 + 255) / 256;

  // bf16 shadow of x
  cvt_bf16_kernel<<<cvt_blocks, 256, 0, stream>>>(x, hbfA, n8);

  const float*  hin   = x;
  const __bf16* hbfin = hbfA;
  for (int l = 0; l < NUM_LAYERS; ++l) {
    // fp32 ping-pong: L0 x->out, L1 out->htmp, L2 htmp->out, L3 out->htmp, L4 htmp->out
    float*  hnext   = (l == NUM_LAYERS - 1) ? out : ((l & 1) ? htmp : out);
    __bf16* hbfnext = (l & 1) ? hbfA : hbfB;

    gemm_msg_kernel<<<gemm_blocks, 256, 0, stream>>>(hbfin, Wt + (size_t)l * D * D, m);
    zero_kernel<<<zero_blocks, 256, 0, stream>>>((float4*)agg, n4);
    edge_scatter_kernel<<<edge_blocks, 256, 0, stream>>>(m, src, dst, eattr, agg);
    cvt_bf16_kernel<<<cvt_blocks, 256, 0, stream>>>(agg, aggbf, n8);
    gru_kernel<<<gru_blocks, 256, 0, stream>>>(aggbf, hbfin, hin, wihb, whhb,
                                               bih, bhh, hnext, hbfnext);

    hin   = hnext;
    hbfin = hbfnext;
  }
}
